// Seq2seq_Model_69526930587903
// MI455X (gfx1250) — matrix operand order repaired and checked by reading, not compiled, not run on hardware
//
#include <hip/hip_runtime.h>

// ---------------- problem constants (from reference) ----------------
#define EMB    8
#define HID    64
#define VOCAB  32000
#define BATCH  32
#define SENC   256
#define SDEC   128
// derived
#define H2     (2*HID)        // 128
#define G4     (4*HID)        // 256  (encoder gates, also FC K)
#define G8     (8*HID)        // 512  (decoder gates)

typedef __bf16 bf16_t;
typedef __attribute__((ext_vector_type(16))) __bf16 v16bf;
typedef __attribute__((ext_vector_type(8)))  __bf16 v8bf;
typedef __attribute__((ext_vector_type(8)))  float  v8f;

__device__ __forceinline__ v8f v8f_zero() {
  v8f z;
#pragma unroll
  for (int i = 0; i < 8; ++i) z[i] = 0.f;
  return z;
}

__device__ __forceinline__ v16bf cat16(v8bf lo, v8bf hi) {
  return __builtin_shufflevector(lo, hi, 0, 1, 2, 3, 4, 5, 6, 7,
                                 8, 9, 10, 11, 12, 13, 14, 15);
}

__device__ __forceinline__ v8f wmma_bf16(v16bf a, v16bf b, v8f c) {
  // D(f32 16x16) = A(16x32 bf16) * B(32x16 bf16) + C
  return __builtin_amdgcn_wmma_f32_16x16x32_bf16(false, a, false, b,
                                                 (short)0, c, false, false);
}

// CDNA5 16-bit A-matrix 16x32 layout: lane (half=lane>>4, m=lane&15) holds
// row m, K-chunks [8*half, 8*half+8) and [16+8*half, 16+8*half+8): two b128
// loads from a row-major [M][K] source at (row, kc).
__device__ __forceinline__ v16bf load_a_frag(const bf16_t* rowk, int half) {
  const v8bf* p = (const v8bf*)rowk;           // 16B-aligned chunks of 8 bf16
  return cat16(p[half], p[2 + half]);
}
// CDNA5 16-bit B-matrix 32x16 layout: lane holds col n=lane&15, K-chunks
// [8*half, 8*half+8) and [16+8*half, 16+8*half+8): two b128 loads from row-major [N][K].
__device__ __forceinline__ v16bf load_bt_frag(const bf16_t* colk, int half) {
  const v8bf* p = (const v8bf*)colk;
  return cat16(p[half], p[2 + half]);
}

__device__ __forceinline__ float sigf(float x) { return 1.f / (1.f + __expf(-x)); }

// ---------------- elementwise helpers ----------------

// strided f32 -> bf16 convert/copy
__global__ void cvt_bf16_kernel(const float* __restrict__ src, int sld,
                                bf16_t* __restrict__ dst, int dld,
                                int rows, int cols) {
  int i = blockIdx.x * blockDim.x + threadIdx.x;
  if (i >= rows * cols) return;
  int r = i / cols, c = i % cols;
  dst[(size_t)r * dld + c] = (bf16_t)src[(size_t)r * sld + c];
}

// transposed f32 -> bf16: dst[c][r] = src[r][c]; src row-major [rows][cols]
__global__ void cvt_bf16_T_kernel(const float* __restrict__ src, long long sStr,
                                  bf16_t* __restrict__ dst, long long dStr,
                                  int rows, int cols) {
  int i = blockIdx.x * blockDim.x + threadIdx.x;
  if (i >= rows * cols) return;
  src += (long long)blockIdx.z * sStr;
  dst += (long long)blockIdx.z * dStr;
  int r = i / cols, c = i % cols;
  dst[(size_t)c * rows + r] = (bf16_t)src[(size_t)r * cols + c];
}

// embedding gather: x[s][b][e] = emb[tok[b][s]][e]
__global__ void embed_kernel(const int* __restrict__ tok, const float* __restrict__ emb,
                             float* __restrict__ x, int S) {
  int i = blockIdx.x * blockDim.x + threadIdx.x;
  if (i >= S * BATCH * EMB) return;
  int e = i % EMB;
  int sb = i / EMB;
  int b = sb % BATCH;
  int s = sb / BATCH;
  x[i] = emb[(size_t)tok[b * S + s] * EMB + e];
}

// input-gate precompute: G[t][b][n] = bias[n] + sum_e x[tt][b][e]*Wih[n][e]
// (K = 8 -> pure VALU; WMMA would be 75% padding waste here)
__global__ void gates_in_kernel(const float* __restrict__ x, const float* __restrict__ Wih,
                                const float* __restrict__ bias, float* __restrict__ G,
                                int T, int fourH, int rev) {
  int i = blockIdx.x * blockDim.x + threadIdx.x;
  if (i >= T * BATCH * fourH) return;
  int n = i % fourH;
  int rb = i / fourH;
  int b = rb % BATCH;
  int t = rb / BATCH;
  int tt = rev ? (T - 1 - t) : t;
  const float* xr = x + ((size_t)tt * BATCH + b) * EMB;
  const float* w  = Wih + (size_t)n * EMB;
  float s = bias[n];
#pragma unroll
  for (int e = 0; e < EMB; ++e) s += xr[e] * w[e];
  G[i] = s;
}

// ---------------- bf16 WMMA GEMM: C[m,n] = sum_k A[m,k]*Bm[n,k] (+bias[n]) ----
// Register-blocked: one wave owns a BMx BN block of 16x16 tiles (32x64 output
// for <2,4>), sharing BM A-fragments across BN B-fragments -> 12 b128 loads
// per 8 WMMAs, ~6 bytes of operand traffic per output byte (all via 192MB L2).
// If Cb != null, output is converted to bf16 and stored there instead of C.
template <int BM, int BN>
__global__ __launch_bounds__(256)
void gemm_bf16_nt_kernel(const bf16_t* __restrict__ A, int lda, long long aStr,
                         const bf16_t* __restrict__ Bm, int ldb, long long bStr,
                         const float* __restrict__ bias,
                         float* __restrict__ C, int ldc, long long cStr,
                         bf16_t* __restrict__ Cb, int ldcb, long long cbStr,
                         int M, int N, int K) {
  const int lane = threadIdx.x & 31;
  const int wave = threadIdx.x >> 5;
  const int half = lane >> 4;
  const int lm   = lane & 15;
  const int mg_count = M / (16 * BM), ng_count = N / (16 * BN);
  const int grp = blockIdx.x * 8 + wave;
  if (grp >= mg_count * ng_count) return;        // wave-uniform; EXEC stays full
  const int mg = grp / ng_count, ng = grp % ng_count;

  A  += (long long)blockIdx.z * aStr;
  Bm += (long long)blockIdx.z * bStr;

  const bf16_t* arow[BM];
  const bf16_t* bcol[BN];
#pragma unroll
  for (int i = 0; i < BM; ++i)
    arow[i] = A + (size_t)(mg * 16 * BM + i * 16 + lm) * lda;
#pragma unroll
  for (int j = 0; j < BN; ++j)
    bcol[j] = Bm + (size_t)(ng * 16 * BN + j * 16 + lm) * ldb;

  v8f acc[BM][BN];
#pragma unroll
  for (int i = 0; i < BM; ++i)
#pragma unroll
    for (int j = 0; j < BN; ++j) acc[i][j] = v8f_zero();

  for (int kc = 0; kc < K; kc += 32) {
    v16bf af[BM];
#pragma unroll
    for (int i = 0; i < BM; ++i) af[i] = load_a_frag(arow[i] + kc, half);
#pragma unroll
    for (int j = 0; j < BN; ++j) {
      v16bf bf = load_bt_frag(bcol[j] + kc, half);
#pragma unroll
      for (int i = 0; i < BM; ++i) acc[i][j] = wmma_bf16(af[i], bf, acc[i][j]);
    }
  }

  if (Cb) {
    Cb += (long long)blockIdx.z * cbStr;
#pragma unroll
    for (int j = 0; j < BN; ++j) {
      const int col = ng * 16 * BN + j * 16 + lm;
      const float bv = bias ? bias[col] : 0.f;
#pragma unroll
      for (int i = 0; i < BM; ++i)
#pragma unroll
        for (int v = 0; v < 8; ++v) {
          int r = mg * 16 * BM + i * 16 + v + 8 * half;  // C layout: M=v+8*half
          Cb[(size_t)r * ldcb + col] = (bf16_t)(acc[i][j][v] + bv);
        }
    }
  } else {
    C += (long long)blockIdx.z * cStr;
#pragma unroll
    for (int j = 0; j < BN; ++j) {
      const int col = ng * 16 * BN + j * 16 + lm;
      const float bv = bias ? bias[col] : 0.f;
#pragma unroll
      for (int i = 0; i < BM; ++i)
#pragma unroll
        for (int v = 0; v < 8; ++v) {
          int r = mg * 16 * BM + i * 16 + v + 8 * half;
          C[(size_t)r * ldc + col] = acc[i][j][v] + bv;
        }
    }
  }
}

// ---------------- persistent LSTM scan ----------------
// One block = whole scan. 8 waves; state h,c (f32) + bf16 shadow of h in LDS.
// Job = (mt in {0,1}) x (jt in 0..H/16-1): wave computes the FOUR gate tiles
// {i,f,g,o} for one (batch-tile, hidden-column-tile) so gate mixing is
// lane-local. Dual param sets so fwd/bwd encoder run as 2 concurrent blocks.
__global__ __launch_bounds__(256)
void lstm_scan_kernel(const float* G_a, const float* G_b,
                      const bf16_t* W_a, const bf16_t* W_b,
                      const float* h0,                       // [32][H] or null
                      float* out, int outSB, int outST, int outOff_a, int outOff_b,
                      int rev_a, int rev_b,
                      float* hT_a, float* hT_b,              // stride 128, or null
                      int H, int T) {
  __shared__ float  h_lds[BATCH * H2];
  __shared__ float  c_lds[BATCH * H2];
  __shared__ alignas(16) bf16_t hbf[BATCH * H2];

  const bool sec = (blockIdx.x != 0);
  const float*  G      = sec ? G_b : G_a;
  const bf16_t* W      = sec ? W_b : W_a;
  const int     outOff = sec ? outOff_b : outOff_a;
  const int     rev    = sec ? rev_b : rev_a;
  float*        hT     = sec ? hT_b : hT_a;

  const int tid = threadIdx.x;
  for (int i = tid; i < BATCH * H; i += 256) {
    float hv = h0 ? h0[i] : 0.f;
    h_lds[i] = hv;
    c_lds[i] = 0.f;
    hbf[i]   = (bf16_t)hv;
  }
  __syncthreads();

  const int lane = tid & 31;
  const int wave = tid >> 5;
  const int half = lane >> 4;
  const int lm   = lane & 15;
  const int jt_count = H >> 4;
  const int njobs    = 2 * jt_count;       // 8 (H=64) or 16 (H=128)
  const int fourH    = 4 * H;

  for (int t = 0; t < T; ++t) {
    float hn[2][8], cn[2][8];

    // ---- compute phase: read prev h (bf16, ds_load_b128) + c; no LDS writes
    for (int js = 0; js < 2; ++js) {
      int job = wave + js * 8;
      if (job >= njobs) break;               // wave-uniform
      int mt = job / jt_count, jt = job % jt_count;
      v8f acc[4];
#pragma unroll
      for (int g = 0; g < 4; ++g) acc[g] = v8f_zero();

      const int arow = mt * 16 + lm;
      for (int kc = 0; kc < H; kc += 32) {
        v16bf a = load_a_frag(&hbf[arow * H + kc], half);
#pragma unroll
        for (int g = 0; g < 4; ++g) {
          const int wrow = g * H + jt * 16 + lm;   // B(k,n) = Whh[n_glob][k]
          v16bf b = load_bt_frag(W + (size_t)wrow * H + kc, half);
          acc[g] = wmma_bf16(a, b, acc[g]);
        }
      }

      const int col = jt * 16 + lm;
#pragma unroll
      for (int v = 0; v < 8; ++v) {
        int brow = mt * 16 + v + 8 * half;
        const float* Gr = G + ((size_t)t * BATCH + brow) * fourH;
        float gi = acc[0][v] + Gr[0 * H + col];
        float gf = acc[1][v] + Gr[1 * H + col];
        float gg = acc[2][v] + Gr[2 * H + col];
        float go = acc[3][v] + Gr[3 * H + col];
        float cold = c_lds[brow * H + col];
        float cnew = sigf(gf) * cold + sigf(gi) * tanhf(gg);
        float hnew = sigf(go) * tanhf(cnew);
        cn[js][v] = cnew;
        hn[js][v] = hnew;
      }
    }

    __syncthreads();   // all reads of h/c done

    // ---- commit phase ----
    const int s = rev ? (T - 1 - t) : t;
    for (int js = 0; js < 2; ++js) {
      int job = wave + js * 8;
      if (job >= njobs) break;
      int mt = job / jt_count, jt = job % jt_count;
      const int col = jt * 16 + lm;
#pragma unroll
      for (int v = 0; v < 8; ++v) {
        int brow = mt * 16 + v + 8 * half;
        float hnew = hn[js][v];
        c_lds[brow * H + col] = cn[js][v];
        h_lds[brow * H + col] = hnew;
        hbf  [brow * H + col] = (bf16_t)hnew;
        out[(size_t)brow * outSB + (size_t)s * outST + outOff + col] = hnew;
      }
    }
    __syncthreads();   // writes visible before next step's reads
  }

  if (hT) {
    for (int i = tid; i < BATCH * H; i += 256) {
      int b = i / H, j = i % H;
      hT[b * H2 + j] = h_lds[i];
    }
  }
}

// ---------------- softmax over s (len 256), one wave per row ----------------
__global__ __launch_bounds__(256)
void softmax_kernel(const float* __restrict__ scores, bf16_t* __restrict__ wgt, int rows) {
  const int lane = threadIdx.x & 31;
  const int wave = threadIdx.x >> 5;
  const int row  = blockIdx.x * 8 + wave;
  if (row >= rows) return;
  const float* sr = scores + (size_t)row * SENC;
  float vals[8];
  float m = -3.402823466e38f;
#pragma unroll
  for (int j = 0; j < 8; ++j) {
    vals[j] = sr[lane + 32 * j];
    m = fmaxf(m, vals[j]);
  }
#pragma unroll
  for (int off = 16; off > 0; off >>= 1) m = fmaxf(m, __shfl_xor(m, off, 32));
  float sum = 0.f;
#pragma unroll
  for (int j = 0; j < 8; ++j) {
    vals[j] = __expf(vals[j] - m);
    sum += vals[j];
  }
#pragma unroll
  for (int off = 16; off > 0; off >>= 1) sum += __shfl_xor(sum, off, 32);
  const float inv = 1.f / sum;
#pragma unroll
  for (int j = 0; j < 8; ++j)
    wgt[(size_t)row * SENC + lane + 32 * j] = (bf16_t)(vals[j] * inv);
}

// ---------------- host orchestration ----------------
extern "C" void kernel_launch(void* const* d_in, const int* in_sizes, int n_in,
                              void* d_out, int out_size, void* d_ws, size_t ws_size,
                              hipStream_t stream) {
  (void)in_sizes; (void)n_in; (void)out_size; (void)ws_size;
  const int*   inp       = (const int*)  d_in[0];
  const int*   dec_inp   = (const int*)  d_in[1];
  const float* emb       = (const float*)d_in[2];
  const float* enc_wih_f = (const float*)d_in[3];
  const float* enc_whh_f = (const float*)d_in[4];
  const float* enc_b_f   = (const float*)d_in[5];
  const float* enc_wih_b = (const float*)d_in[6];
  const float* enc_whh_b = (const float*)d_in[7];
  const float* enc_b_b   = (const float*)d_in[8];
  const float* dec_wih   = (const float*)d_in[9];
  const float* dec_whh   = (const float*)d_in[10];
  const float* dec_b     = (const float*)d_in[11];
  const float* att       = (const float*)d_in[12];
  const float* fc_w      = (const float*)d_in[13];
  const float* fc_b      = (const float*)d_in[14];
  float* out = (float*)d_out;

  // bump allocator over d_ws (~62 MB total)
  char* ws = (char*)d_ws;
  size_t off = 0;
  auto alloc = [&](size_t elems, size_t esz) -> void* {
    void* p = ws + off;
    off = (off + elems * esz + 255) & ~(size_t)255;
    return p;
  };

  float* x_enc   = (float*)alloc((size_t)SENC * BATCH * EMB, 4);
  float* x_dec   = (float*)alloc((size_t)SDEC * BATCH * EMB, 4);
  float* Gf      = (float*)alloc((size_t)SENC * BATCH * G4, 4);
  float* Gb      = (float*)alloc((size_t)SENC * BATCH * G4, 4);
  float* Gd      = (float*)alloc((size_t)SDEC * BATCH * G8, 4);
  float* out_enc = (float*)alloc((size_t)BATCH * SENC * H2, 4);
  float* out_dec = (float*)alloc((size_t)BATCH * SDEC * H2, 4);
  float* h0dec   = (float*)alloc((size_t)BATCH * H2, 4);
  float* scores  = (float*)alloc((size_t)BATCH * SDEC * SENC, 4);

  bf16_t* whhf_bf = (bf16_t*)alloc((size_t)G4 * HID, 2);
  bf16_t* whhb_bf = (bf16_t*)alloc((size_t)G4 * HID, 2);
  bf16_t* whhd_bf = (bf16_t*)alloc((size_t)G8 * H2, 2);
  bf16_t* attT_bf = (bf16_t*)alloc((size_t)H2 * H2, 2);            // att^T [e][d]
  bf16_t* fcw_bf  = (bf16_t*)alloc((size_t)VOCAB * G4, 2);
  bf16_t* enc_bf  = (bf16_t*)alloc((size_t)BATCH * SENC * H2, 2);  // [b][s][d]
  bf16_t* encT_bf = (bf16_t*)alloc((size_t)BATCH * H2 * SENC, 2);  // [b][d][s]
  bf16_t* adec_bf = (bf16_t*)alloc((size_t)BATCH * SDEC * H2, 2);
  bf16_t* wgt_bf  = (bf16_t*)alloc((size_t)BATCH * SDEC * SENC, 2);
  bf16_t* cat_bf  = (bf16_t*)alloc((size_t)BATCH * SDEC * G4, 2);  // [4096][256]

  const int THR = 256;
  auto cdiv = [](long long a, long long b) { return (int)((a + b - 1) / b); };
  // wave-groups for the <2,4> GEMM: each wave does a 32x64 output block
  auto gemm_grid = [&](int M, int N) { return cdiv((long long)(M / 32) * (N / 64), 8); };

  // 1) weight conversions to bf16 (att transposed so GEMMs are pure A*B^T)
  cvt_bf16_kernel<<<cdiv((long long)G4 * HID, THR), THR, 0, stream>>>(enc_whh_f, HID, whhf_bf, HID, G4, HID);
  cvt_bf16_kernel<<<cdiv((long long)G4 * HID, THR), THR, 0, stream>>>(enc_whh_b, HID, whhb_bf, HID, G4, HID);
  cvt_bf16_kernel<<<cdiv((long long)G8 * H2, THR), THR, 0, stream>>>(dec_whh, H2, whhd_bf, H2, G8, H2);
  cvt_bf16_T_kernel<<<dim3(cdiv((long long)H2 * H2, THR), 1, 1), THR, 0, stream>>>(att, 0, attT_bf, 0, H2, H2);
  cvt_bf16_kernel<<<cdiv((long long)VOCAB * G4, THR), THR, 0, stream>>>(fc_w, G4, fcw_bf, G4, VOCAB, G4);

  // 2) embeddings
  embed_kernel<<<cdiv((long long)SENC * BATCH * EMB, THR), THR, 0, stream>>>(inp, emb, x_enc, SENC);
  embed_kernel<<<cdiv((long long)SDEC * BATCH * EMB, THR), THR, 0, stream>>>(dec_inp, emb, x_dec, SDEC);

  // 3) time-independent input projections (+combined bias)
  gates_in_kernel<<<cdiv((long long)SENC * BATCH * G4, THR), THR, 0, stream>>>(x_enc, enc_wih_f, enc_b_f, Gf, SENC, G4, 0);
  gates_in_kernel<<<cdiv((long long)SENC * BATCH * G4, THR), THR, 0, stream>>>(x_enc, enc_wih_b, enc_b_b, Gb, SENC, G4, 1);
  gates_in_kernel<<<cdiv((long long)SDEC * BATCH * G8, THR), THR, 0, stream>>>(x_dec, dec_wih, dec_b, Gd, SDEC, G8, 0);

  // 4) encoder bi-LSTM: fwd + bwd as 2 concurrent persistent blocks
  lstm_scan_kernel<<<2, 256, 0, stream>>>(
      Gf, Gb, whhf_bf, whhb_bf, /*h0=*/nullptr,
      out_enc, SENC * H2, H2, /*off fwd*/0, /*off bwd*/HID,
      /*rev*/0, 1, /*hT*/h0dec, h0dec + HID, HID, SENC);

  // 5) decoder LSTM (hidden 128), h0 = [hf|hb], c0 = 0
  lstm_scan_kernel<<<1, 256, 0, stream>>>(
      Gd, Gd, whhd_bf, whhd_bf, h0dec,
      out_dec, SDEC * H2, H2, 0, 0, 0, 0, nullptr, nullptr, H2, SDEC);

  // 6) bf16 copies: out_dec -> cat[:, :128]; out_enc -> enc_bf and encT_bf
  cvt_bf16_kernel<<<cdiv((long long)BATCH * SDEC * H2, THR), THR, 0, stream>>>(
      out_dec, H2, cat_bf, G4, BATCH * SDEC, H2);
  cvt_bf16_kernel<<<cdiv((long long)BATCH * SENC * H2, THR), THR, 0, stream>>>(
      out_enc, H2, enc_bf, H2, BATCH * SENC, H2);
  cvt_bf16_T_kernel<<<dim3(cdiv((long long)SENC * H2, THR), 1, BATCH), THR, 0, stream>>>(
      out_enc, (long long)SENC * H2, encT_bf, (long long)H2 * SENC, SENC, H2);

  // 7) Adec = out_dec @ att = cat[:,:128] * attT^T  -> bf16 directly
  gemm_bf16_nt_kernel<2, 4><<<dim3(gemm_grid(BATCH * SDEC, H2), 1, 1), 256, 0, stream>>>(
      cat_bf, G4, 0, attT_bf, H2, 0, nullptr,
      nullptr, 0, 0, adec_bf, H2, 0,
      BATCH * SDEC, H2, H2);

  // 8) scores[b] = Adec_b * enc_b^T   (batched over 32) -> f32 for softmax
  gemm_bf16_nt_kernel<2, 4><<<dim3(gemm_grid(SDEC, SENC), 1, BATCH), 256, 0, stream>>>(
      adec_bf, H2, (long long)SDEC * H2,
      enc_bf, H2, (long long)SENC * H2, nullptr,
      scores, SENC, (long long)SDEC * SENC, nullptr, 0, 0,
      SDEC, SENC, H2);

  // 9) softmax over s -> bf16 weights
  softmax_kernel<<<cdiv((long long)BATCH * SDEC, 8), 256, 0, stream>>>(scores, wgt_bf, BATCH * SDEC);

  // 10) ctx[b] = wgt_b * encT_b^T  (batched) -> bf16 straight into cat[:,128:]
  gemm_bf16_nt_kernel<2, 4><<<dim3(gemm_grid(SDEC, H2), 1, BATCH), 256, 0, stream>>>(
      wgt_bf, SENC, (long long)SDEC * SENC,
      encT_bf, SENC, (long long)H2 * SENC, nullptr,
      nullptr, 0, 0, cat_bf + H2, G4, (long long)SDEC * G4,
      SDEC, H2, SENC);

  // 11) FC: out = cat * fc_w^T + fc_b   [4096,32000] — output-BW bound
  gemm_bf16_nt_kernel<2, 4><<<dim3(gemm_grid(BATCH * SDEC, VOCAB), 1, 1), 256, 0, stream>>>(
      cat_bf, G4, 0, fcw_bf, G4, 0, fc_b,
      out, VOCAB, 0, nullptr, 0, 0,
      BATCH * SDEC, VOCAB, G4);
}
